// Attention_19421842113041
// MI455X (gfx1250) — matrix operand order repaired and checked by reading, not compiled, not run on hardware
//
#include <hip/hip_runtime.h>

// ---------------------------------------------------------------------------
// Attention pipeline for MI455X (gfx1250, wave32, WMMA 16x16x32 f16).
//   B=2, N=2048, HEADS=16, DIM_HEAD=64, QUERY_DIM=INNER=1024
// All GEMMs via v_wmma_f32_16x16x32_f16 (f16 in, f32 accumulate).
// Tile staging via GLOBAL_LOAD_ASYNC_TO_LDS_B128 (ASYNCcnt) into ping-pong
// LDS buffers: one barrier per K-step, zero VGPR staging.
// ---------------------------------------------------------------------------

typedef __attribute__((ext_vector_type(16))) _Float16     v16h;
typedef __attribute__((ext_vector_type(4)))  _Float16     v4h;
typedef __attribute__((ext_vector_type(8)))  float        v8f;
typedef __attribute__((ext_vector_type(4)))  unsigned int v4u;

union F16x16 { v16h h; v4u u[2]; };
union F16x8  { v4u u; _Float16 h[8]; };

__device__ __forceinline__ v8f wmma_f16(v16h a, v16h b, v8f c) {
  // 8 args: (neg_a, A, neg_b, B, c_mod, C, reuse_a, reuse_b)
  return __builtin_amdgcn_wmma_f32_16x16x32_f16(false, a, false, b,
                                                (short)0, c, false, false);
}

// ---- CDNA5 async global->LDS copy (ASYNCcnt). VDST = per-lane LDS byte
// offset, VADDR = 64-bit global address (GV mode, saddr off). ----
__device__ __forceinline__ void async_b128(unsigned lds_byte_off, const void* g) {
  asm volatile("global_load_async_to_lds_b128 %0, %1, off"
               :: "v"(lds_byte_off), "v"((unsigned long long)g)
               : "memory");
}
__device__ __forceinline__ void wait_async0() {
  asm volatile("s_wait_asynccnt 0x0" ::: "memory");
}
// Generic pointer to LDS keeps the LDS byte offset in addr[31:0].
__device__ __forceinline__ unsigned lds_off(const void* p) {
  return (unsigned)(unsigned long long)p;
}

// ------------------------------- f32 -> f16 --------------------------------
__global__ void cvt_f32_f16(const float* __restrict__ src,
                            _Float16* __restrict__ dst, int n4) {
  int i = blockIdx.x * blockDim.x + threadIdx.x;
  if (i >= n4) return;
  float4 f = ((const float4*)src)[i];
  v4h h;
  h.x = (_Float16)f.x; h.y = (_Float16)f.y;
  h.z = (_Float16)f.z; h.w = (_Float16)f.w;
  ((v4h*)dst)[i] = h;
}

// --------------------- f32 -> f16 transpose (weights) ----------------------
// src: (R x C) f32 row-major  ->  dst: (C x R) f16 row-major.
__global__ __launch_bounds__(256) void cvt_transpose(
    const float* __restrict__ src, _Float16* __restrict__ dst, int R, int C) {
  __shared__ _Float16 tile[32][33];
  const int r0 = blockIdx.y * 32;
  const int c0 = blockIdx.x * 32;
  const int tr = threadIdx.x >> 3;        // 0..31
  const int tc = (threadIdx.x & 7) * 4;   // 0,4,...,28

  float4 f = *(const float4*)&src[(size_t)(r0 + tr) * C + c0 + tc];
  tile[tr][tc + 0] = (_Float16)f.x;
  tile[tr][tc + 1] = (_Float16)f.y;
  tile[tr][tc + 2] = (_Float16)f.z;
  tile[tr][tc + 3] = (_Float16)f.w;
  __syncthreads();

  v4h h;
  h.x = tile[tc + 0][tr];
  h.y = tile[tc + 1][tr];
  h.z = tile[tc + 2][tr];
  h.w = tile[tc + 3][tr];
  *(v4h*)&dst[(size_t)(c0 + tr) * R + r0 + tc] = h;
}

// ------------------------- shared WMMA GEMM mainloop -----------------------
// 256 threads = 8 waves in 2(M) x 4(N); per-wave 32x64; WG tile 64x256.
// K-step 32. Ping-pong LDS buffers filled by async global->LDS b128 copies:
// per iteration: s_wait_asynccnt 0 -> barrier -> issue next tile -> WMMA.
// Ash: 2 x (64 rows x stride 40 halves). Bsh: 2 x (256 cols x stride 40).
__device__ __forceinline__ void gemm_mainloop_async(
    const _Float16* __restrict__ A, int lda,
    const _Float16* __restrict__ Bt,   // (Ncols x K) row-major (pre-transposed)
    int gm0, int cB0, int K,
    _Float16* Ash, _Float16* Bsh, v8f acc[2][4])
{
  const int t = threadIdx.x;
  const int lane = t & 31;
  const int r  = lane & 15;
  const int hi = lane >> 4;
  const int wv = t >> 5;
  const int wm = (wv & 1) * 32;
  const int wn = (wv >> 1) * 64;

  const int arow = t >> 2, aseg = t & 3;       // A fill: 64 rows x (4 x 8 h)
  const _Float16* aptr = A  + (size_t)(gm0 + arow) * lda + aseg * 8;
  const _Float16* bptr = Bt + (size_t)(cB0 + t) * K;   // one n-column / thread

  const unsigned ABUF = 64 * 40 * 2;           // bytes per A buffer
  const unsigned BBUF = 256 * 40 * 2;          // bytes per B buffer
  const unsigned aoff0 = lds_off(Ash) + (unsigned)(arow * 40 + aseg * 8) * 2;
  const unsigned boff0 = lds_off(Bsh) + (unsigned)t * 80;

  // ---- prologue: async-copy k-tile 0 into buffer 0 ----
  async_b128(aoff0,      aptr);
  async_b128(boff0 +  0, bptr +  0);
  async_b128(boff0 + 16, bptr +  8);
  async_b128(boff0 + 32, bptr + 16);
  async_b128(boff0 + 48, bptr + 24);

  int buf = 0;
  for (int k0 = 0; k0 < K; k0 += 32, buf ^= 1) {
    wait_async0();        // own async copies for current buffer landed
    __syncthreads();      // everyone's landed; other buffer no longer read

    if (k0 + 32 < K) {    // issue next tile into the other buffer
      const unsigned ao = aoff0 + (unsigned)(buf ^ 1) * ABUF;
      const unsigned bo = boff0 + (unsigned)(buf ^ 1) * BBUF;
      const _Float16* ap = aptr + k0 + 32;
      const _Float16* bp = bptr + k0 + 32;
      async_b128(ao,      ap);
      async_b128(bo +  0, bp +  0);
      async_b128(bo + 16, bp +  8);
      async_b128(bo + 32, bp + 16);
      async_b128(bo + 48, bp + 24);
    }

    const _Float16* AshC = Ash + (size_t)buf * (64 * 40);
    const _Float16* BshC = Bsh + (size_t)buf * (256 * 40);

    F16x16 af[2], bf[4];
#pragma unroll
    for (int mi = 0; mi < 2; ++mi) {
      const _Float16* ap = &AshC[(wm + mi * 16 + r) * 40 + hi * 8];
      af[mi].u[0] = *(const v4u*)ap;          // K = hi*8 .. +7
      af[mi].u[1] = *(const v4u*)(ap + 16);   // K = hi*8+16 .. +23
    }
#pragma unroll
    for (int ni = 0; ni < 4; ++ni) {
      const _Float16* bp = &BshC[(wn + ni * 16 + r) * 40 + hi * 8];
      bf[ni].u[0] = *(const v4u*)bp;          // K = hi*8 .. +7
      bf[ni].u[1] = *(const v4u*)(bp + 16);   // K = hi*8+16 .. +23
    }
#pragma unroll
    for (int mi = 0; mi < 2; ++mi)
#pragma unroll
      for (int ni = 0; ni < 4; ++ni)
        acc[mi][ni] = wmma_f16(af[mi].h, bf[ni].h, acc[mi][ni]);
  }
}

// ------------------------------ QKV projection -----------------------------
__global__ __launch_bounds__(256) void qkv_gemm(
    const _Float16* __restrict__ xh,
    const _Float16* __restrict__ WqT,    // (1024 x 1024)
    const _Float16* __restrict__ WkvT,   // (2048 x 1024)
    _Float16* __restrict__ qh, _Float16* __restrict__ kh,
    _Float16* __restrict__ vT)
{
  __shared__ _Float16 Ash[2 * 64 * 40];
  __shared__ _Float16 Bsh[2 * 256 * 40];

  const int gm0 = blockIdx.x * 64;
  const int c0  = blockIdx.y * 256;
  const int region = c0 >> 10;            // 0=q, 1=k, 2=v

  const _Float16* Bt; int cB0;
  if (region == 0) { Bt = WqT;  cB0 = c0; }
  else             { Bt = WkvT; cB0 = c0 - 1024; }

  v8f zero = {0.f,0.f,0.f,0.f,0.f,0.f,0.f,0.f};
  v8f acc[2][4];
#pragma unroll
  for (int i = 0; i < 2; ++i)
#pragma unroll
    for (int j = 0; j < 4; ++j) acc[i][j] = zero;

  gemm_mainloop_async(xh, 1024, Bt, gm0, cB0, 1024, Ash, Bsh, acc);

  const int t = threadIdx.x, lane = t & 31, wv = t >> 5;
  const int r = lane & 15, hi = lane >> 4;
  const int wm = (wv & 1) * 32, wn = (wv >> 1) * 64;

  if (region == 2) {
    // v: pack 8 consecutive n-rows per lane -> one b128 store into (B,H,D,N)
#pragma unroll
    for (int mi = 0; mi < 2; ++mi)
#pragma unroll
      for (int ni = 0; ni < 4; ++ni) {
        int col = c0 + wn + ni * 16 + r - 2048;        // 0..1023
        int hh = col >> 6, dd = col & 63;
        int rowbase = gm0 + wm + mi * 16 + hi * 8;     // 8 consecutive rows
        int b = rowbase >> 11, nn = rowbase & 2047;
        F16x8 pk;
#pragma unroll
        for (int v = 0; v < 8; ++v) pk.h[v] = (_Float16)acc[mi][ni][v];
        *(v4u*)&vT[((size_t)(b * 16 + hh) * 64 + dd) * 2048 + nn] = pk.u;
      }
  } else {
    _Float16* dst = (region == 0) ? qh : kh;
    const float scl = (region == 0) ? 0.125f : 1.f;    // fold DIM_HEAD^-0.5
    const int cb = (region == 0) ? c0 : c0 - 1024;
#pragma unroll
    for (int mi = 0; mi < 2; ++mi)
#pragma unroll
      for (int ni = 0; ni < 4; ++ni)
#pragma unroll
        for (int v = 0; v < 8; ++v) {
          int row = gm0 + wm + mi * 16 + v + hi * 8;   // b*2048 + n
          int col = cb + wn + ni * 16 + r;             // 0..1023
          int b = row >> 11, nn = row & 2047;
          int hh = col >> 6, dd = col & 63;
          dst[((size_t)(b * 16 + hh) * 2048 + nn) * 64 + dd] =
              (_Float16)(acc[mi][ni][v] * scl);
        }
  }
}

// ------------------------------ flash attention ----------------------------
// grid = (N/128, B*H). 8 waves/WG; wave owns a 16-row q block. 64-key/value
// tiles async-copied into ping-pong LDS buffers; online softmax in registers.
__global__ __launch_bounds__(256) void attn_kernel(
    const _Float16* __restrict__ qh, const _Float16* __restrict__ kh,
    const _Float16* __restrict__ vT, _Float16* __restrict__ Zh)
{
  __shared__ _Float16 Ksh[2 * 64 * 72];    // [j][d] row-major  (B of Q@K^T)
  __shared__ _Float16 Vsh[2 * 64 * 72];    // [d][j]            (B of P@V)
  __shared__ _Float16 Psh[8 * 16 * 72];    // per-wave P scratch (16 x 64)

  const int bh = blockIdx.y;
  const int t = threadIdx.x, lane = t & 31, wv = t >> 5;
  const int r = lane & 15, hi = lane >> 4;
  const int i0 = blockIdx.x * 128 + wv * 16;

  // Q fragments (A layout), 2 K-chunks of 32 over d=64, loaded once.
  F16x16 qf[2];
  {
    const _Float16* qp = qh + ((size_t)bh * 2048 + i0 + r) * 64;
#pragma unroll
    for (int c = 0; c < 2; ++c) {
      const _Float16* p = qp + c * 32 + hi * 8;
      qf[c].u[0] = *(const v4u*)p;
      qf[c].u[1] = *(const v4u*)(p + 16);
    }
  }

  v8f zero = {0.f,0.f,0.f,0.f,0.f,0.f,0.f,0.f};
  v8f acc_o[4];
#pragma unroll
  for (int dt = 0; dt < 4; ++dt) acc_o[dt] = zero;
  float mrow[8], srow[8];
#pragma unroll
  for (int v = 0; v < 8; ++v) { mrow[v] = -3.0e38f; srow[v] = 0.f; }

  const int fj = t >> 2, fseg = t & 3;     // K: row j / V: row d; 4x16 halves
  const _Float16* kgp = kh + ((size_t)bh * 2048 + fj) * 64 + fseg * 16;
  const _Float16* vgp = vT + ((size_t)bh * 64 + fj) * 2048 + fseg * 16;

  const unsigned KVBUF = 64 * 72 * 2;      // bytes per buffer
  const unsigned koff0 = lds_off(Ksh) + (unsigned)(fj * 72 + fseg * 16) * 2;
  const unsigned voff0 = lds_off(Vsh) + (unsigned)(fj * 72 + fseg * 16) * 2;

  // ---- prologue: async-copy tile j0=0 into buffer 0 ----
  async_b128(koff0,      kgp);
  async_b128(koff0 + 16, kgp + 8);
  async_b128(voff0,      vgp);
  async_b128(voff0 + 16, vgp + 8);

  int buf = 0;
  for (int j0 = 0; j0 < 2048; j0 += 64, buf ^= 1) {
    wait_async0();
    __syncthreads();

    if (j0 + 64 < 2048) {                  // issue next tile into other buffer
      const unsigned ko = koff0 + (unsigned)(buf ^ 1) * KVBUF;
      const unsigned vo = voff0 + (unsigned)(buf ^ 1) * KVBUF;
      const _Float16* kp = kgp + (size_t)(j0 + 64) * 64;
      const _Float16* vp = vgp + (j0 + 64);
      async_b128(ko,      kp);
      async_b128(ko + 16, kp + 8);
      async_b128(vo,      vp);
      async_b128(vo + 16, vp + 8);
    }

    const _Float16* KshC = Ksh + (size_t)buf * (64 * 72);
    const _Float16* VshC = Vsh + (size_t)buf * (64 * 72);

    // ---- sim tiles: 4 x (16x16), K-dim = d (2 chunks of 32) ----
    v8f sacc[4];
#pragma unroll
    for (int jt = 0; jt < 4; ++jt) {
      sacc[jt] = zero;
#pragma unroll
      for (int c = 0; c < 2; ++c) {
        F16x16 kb;
        const _Float16* kp = &KshC[(jt * 16 + r) * 72 + c * 32 + hi * 8];
        kb.u[0] = *(const v4u*)kp;
        kb.u[1] = *(const v4u*)(kp + 16);
        sacc[jt] = wmma_f16(qf[c].h, kb.h, sacc[jt]);
      }
    }

    // ---- online softmax (row = v + hi*8, spread over 16 lanes) ----
#pragma unroll
    for (int v = 0; v < 8; ++v) {
      float mx = fmaxf(fmaxf(sacc[0][v], sacc[1][v]),
                       fmaxf(sacc[2][v], sacc[3][v]));
#pragma unroll
      for (int off = 1; off < 16; off <<= 1)
        mx = fmaxf(mx, __shfl_xor(mx, off, 32));
      float mnew  = fmaxf(mrow[v], mx);
      float alpha = __expf(mrow[v] - mnew);
      float rs = 0.f;
#pragma unroll
      for (int jt = 0; jt < 4; ++jt) {
        float p = __expf(sacc[jt][v] - mnew);
        sacc[jt][v] = p;
        rs += p;
      }
#pragma unroll
      for (int off = 1; off < 16; off <<= 1)
        rs += __shfl_xor(rs, off, 32);
      srow[v] = srow[v] * alpha + rs;
      mrow[v] = mnew;
#pragma unroll
      for (int dt = 0; dt < 4; ++dt) acc_o[dt][v] *= alpha;
    }

    // ---- C-layout -> A-layout for P via per-wave LDS scratch ----
    _Float16* pbase = &Psh[wv * 16 * 72];
#pragma unroll
    for (int jt = 0; jt < 4; ++jt)
#pragma unroll
      for (int v = 0; v < 8; ++v)
        pbase[(v + hi * 8) * 72 + jt * 16 + r] = (_Float16)sacc[jt][v];

    F16x16 pf[2];   // same-wave LDS RAW: DS stays in order (s_wait_dscnt)
#pragma unroll
    for (int c = 0; c < 2; ++c) {
      const _Float16* pp = &pbase[r * 72 + c * 32 + hi * 8];
      pf[c].u[0] = *(const v4u*)pp;
      pf[c].u[1] = *(const v4u*)(pp + 16);
    }

    // ---- PV: acc_o[dt] += P(16x64) @ V(64x16) ----
#pragma unroll
    for (int dt = 0; dt < 4; ++dt)
#pragma unroll
      for (int c = 0; c < 2; ++c) {
        F16x16 vb;
        const _Float16* vp = &VshC[(dt * 16 + r) * 72 + c * 32 + hi * 8];
        vb.u[0] = *(const v4u*)vp;
        vb.u[1] = *(const v4u*)(vp + 16);
        acc_o[dt] = wmma_f16(pf[c].h, vb.h, acc_o[dt]);
      }
  }

  // ---- normalize + store into Z with quirky rearrange: n -> (hh=n>>7, m) --
#pragma unroll
  for (int dt = 0; dt < 4; ++dt)
#pragma unroll
    for (int v = 0; v < 8; ++v) {
      int n  = i0 + v + hi * 8;
      int hh = n >> 7, mr = n & 127;
      int d  = dt * 16 + r;
      float val = acc_o[dt][v] / srow[v];
      Zh[((size_t)bh * 128 + mr) * 1024 + hh * 64 + d] = (_Float16)val;
    }
}

// ----------------------------- output projection ---------------------------
__global__ __launch_bounds__(256) void out_gemm(
    const _Float16* __restrict__ Zh, const _Float16* __restrict__ WoT,
    const float* __restrict__ bo, float* __restrict__ out)
{
  __shared__ _Float16 Ash[2 * 64 * 40];
  __shared__ _Float16 Bsh[2 * 256 * 40];

  const int gm0 = blockIdx.x * 64;
  const int c0  = blockIdx.y * 256;

  v8f zero = {0.f,0.f,0.f,0.f,0.f,0.f,0.f,0.f};
  v8f acc[2][4];
#pragma unroll
  for (int i = 0; i < 2; ++i)
#pragma unroll
    for (int j = 0; j < 4; ++j) acc[i][j] = zero;

  gemm_mainloop_async(Zh, 1024, WoT, gm0, c0, 1024, Ash, Bsh, acc);

  const int t = threadIdx.x, lane = t & 31, wv = t >> 5;
  const int r = lane & 15, hi = lane >> 4;
  const int wm = (wv & 1) * 32, wn = (wv >> 1) * 64;

#pragma unroll
  for (int mi = 0; mi < 2; ++mi)
#pragma unroll
    for (int ni = 0; ni < 4; ++ni)
#pragma unroll
      for (int v = 0; v < 8; ++v) {
        int row = gm0 + wm + mi * 16 + v + hi * 8;
        int col = c0  + wn + ni * 16 + r;
        out[(size_t)row * 1024 + col] = acc[mi][ni][v] + bo[col];
      }
}

// --------------------------------- launcher --------------------------------
extern "C" void kernel_launch(void* const* d_in, const int* in_sizes, int n_in,
                              void* d_out, int out_size, void* d_ws, size_t ws_size,
                              hipStream_t stream)
{
  const float* x   = (const float*)d_in[0];   // (2, 2048, 1024)
  const float* Wq  = (const float*)d_in[1];   // (1024, 1024)
  const float* Wkv = (const float*)d_in[2];   // (1024, 2048)
  const float* Wo  = (const float*)d_in[3];   // (1024, 1024)
  const float* bo  = (const float*)d_in[4];   // (1024,)
  float* out = (float*)d_out;                 // (32, 128, 1024)

  char* ws = (char*)d_ws;                     // ~48 MB used
  _Float16* xh   = (_Float16*)(ws);                        // 8 MB (4096,1024)
  _Float16* WqT  = (_Float16*)(ws + (size_t)( 8u << 20));  // 2 MB (1024,1024)
  _Float16* WkvT = (_Float16*)(ws + (size_t)(10u << 20));  // 4 MB (2048,1024)
  _Float16* WoT  = (_Float16*)(ws + (size_t)(14u << 20));  // 2 MB (1024,1024)
  _Float16* qh   = (_Float16*)(ws + (size_t)(16u << 20));  // 8 MB (B,H,N,D)
  _Float16* kh   = (_Float16*)(ws + (size_t)(24u << 20));  // 8 MB (B,H,N,D)
  _Float16* vT   = (_Float16*)(ws + (size_t)(32u << 20));  // 8 MB (B,H,D,N)
  _Float16* Zh   = (_Float16*)(ws + (size_t)(40u << 20));  // 8 MB (32,128,1024)

  cvt_f32_f16  <<<4096, 256, 0, stream>>>(x, xh, 1048576);
  cvt_transpose<<<dim3(32, 32), 256, 0, stream>>>(Wq,  WqT,  1024, 1024);
  cvt_transpose<<<dim3(64, 32), 256, 0, stream>>>(Wkv, WkvT, 1024, 2048);
  cvt_transpose<<<dim3(32, 32), 256, 0, stream>>>(Wo,  WoT,  1024, 1024);

  qkv_gemm   <<<dim3(64, 12), 256, 0, stream>>>(xh, WqT, WkvT, qh, kh, vT);
  attn_kernel<<<dim3(16, 32), 256, 0, stream>>>(qh, kh, vT, Zh);
  out_gemm   <<<dim3(64, 4),  256, 0, stream>>>(Zh, WoT, bo, out);
}
